// MultiHeadAttentionPE_50551765074015
// MI455X (gfx1250) — matrix operand order repaired and checked by reading, not compiled, not run on hardware
//
#include <hip/hip_runtime.h>

typedef __attribute__((ext_vector_type(16))) __bf16 v16bf;
typedef __attribute__((ext_vector_type(8)))  __bf16 v8bf;
typedef __attribute__((ext_vector_type(8)))  float  v8f;
typedef __attribute__((ext_vector_type(8)))  float  v8ff;

#define Bsz 4
#define SEQ 1024
#define DM  768
#define NH  12
#define DH  64
#define MTOT (Bsz*SEQ)   // 4096
#define QT  64           // queries per attention block

// ---------------------------------------------------------------------------
// WMMA fragment helpers (ISA 7.12.2, wave32), contiguous-per-lane layouts:
//  A (16x32 bf16, row-major, ld elems): lane(h=lane>>4,m=lane&15) reads
//    A[m][16h..16h+7] and A[m][16h+8..+7]  -> two 16B LDS loads
//  B stored TRANSPOSED Bt[n][k] (ld elems): lane reads Bt[n][16h..16h+15]
//    -> one 32B LDS load
//  C/D (16x16 f32): vgpr r -> (m = r + 8*half, n = lane&15)
// ---------------------------------------------------------------------------
__device__ inline v8f wmma_bf16(v16bf a, v16bf b, v8f c) {
  return __builtin_amdgcn_wmma_f32_16x16x32_bf16(false, a, false, b, (short)0, c,
                                                 false, false);
}

__device__ inline v16bf load_a_bf(const __bf16* __restrict__ s, int ld) {
  int lane = threadIdx.x & 31;
  int half = lane >> 4, m = lane & 15;
  const __bf16* p = s + m * ld + 16 * half;
  v8bf lo = *(const v8bf*)(p);
  v8bf hi = *(const v8bf*)(p + 8);
  return __builtin_shufflevector(lo, hi, 0, 1, 2, 3, 4, 5, 6, 7, 8, 9, 10, 11,
                                 12, 13, 14, 15);
}

__device__ inline v16bf load_a_f32(const float* __restrict__ s, int ld) {
  int lane = threadIdx.x & 31;
  int half = lane >> 4, m = lane & 15;
  const float* p = s + m * ld + 16 * half;
  v8ff lo = *(const v8ff*)(p);
  v8ff hi = *(const v8ff*)(p + 8);
  v16bf a;
#pragma unroll
  for (int i = 0; i < 8; ++i) {
    a[i]     = (__bf16)lo[i];
    a[i + 8] = (__bf16)hi[i];
  }
  return a;
}

__device__ inline v16bf load_bt(const __bf16* __restrict__ s, int ld) {
  int lane = threadIdx.x & 31;
  int half = lane >> 4, n = lane & 15;
  return *(const v16bf*)(s + n * ld + 16 * half);
}

// ---------------------------------------------------------------------------
// GEMM: Out = (A[M,K] @ W[Nc,K]^T + bias) * scale
// AMODE: 0 = fp32 A, 1 = bf16 A
// OMODE: 0 = bf16 row-major [M,Nc], 1 = fp32 row-major,
//        2 = bf16 head-transposed [B,NH,DH,SEQ]   (for V)
// Block tile 128x64, k-step 32, 8 waves; wave -> 32x32 register tile
// (4 WMMAs / k-step from 2 A-frags + 2 B-frags: 2x reuse on both operands).
// ---------------------------------------------------------------------------
template <int AMODE, int OMODE>
__global__ __launch_bounds__(256) void gemm_bt_kernel(
    const void* __restrict__ Ap, const float* __restrict__ W,
    const float* __restrict__ bias, float scale, void* __restrict__ Outp,
    int M, int Nc, int K) {
  __shared__ __attribute__((aligned(32))) __bf16 As[128 * 32];  // [m][k]
  __shared__ __attribute__((aligned(32))) __bf16 Bs[64 * 32];   // Bt: [n][k]

  const int tid  = threadIdx.x;
  const int lane = tid & 31;
  const int wave = tid >> 5;
  const int m0 = blockIdx.x * 128;
  const int n0 = blockIdx.y * 64;
  const int mr = (wave >> 1) * 32;  // 0,32,64,96
  const int nc = (wave & 1) * 32;   // 0,32

  v8f c00 = {}, c01 = {}, c10 = {}, c11 = {};

  const int arow = tid >> 1, acb = (tid & 1) * 16;  // A staging: 16 elems/thread
  const int brow = tid >> 2, bcb = (tid & 3) * 8;   // B staging: 8 elems/thread

  for (int k0 = 0; k0 < K; k0 += 32) {
    __syncthreads();
    // ---- stage A tile [128m][32k] ----
    if (AMODE == 0) {
      const float* A = (const float*)Ap;
      const float* ap = A + (size_t)(m0 + arow) * K + k0 + acb;
      v8ff lo = *(const v8ff*)(ap);
      v8ff hi = *(const v8ff*)(ap + 8);
      v16bf ab;
#pragma unroll
      for (int i = 0; i < 8; ++i) {
        ab[i]     = (__bf16)lo[i];
        ab[i + 8] = (__bf16)hi[i];
      }
      *(v16bf*)(As + arow * 32 + acb) = ab;
    } else {
      const __bf16* A = (const __bf16*)Ap;
      *(v16bf*)(As + arow * 32 + acb) =
          *(const v16bf*)(A + (size_t)(m0 + arow) * K + k0 + acb);
    }
    // ---- stage Bt tile [64n][32k] ----
    {
      v8ff wv8 = *(const v8ff*)(W + (size_t)(n0 + brow) * K + k0 + bcb);
      v8bf wb;
#pragma unroll
      for (int i = 0; i < 8; ++i) wb[i] = (__bf16)wv8[i];
      *(v8bf*)(Bs + brow * 32 + bcb) = wb;
    }
    __syncthreads();
    v16bf a0 = load_a_bf(As + mr * 32,        32);
    v16bf a1 = load_a_bf(As + (mr + 16) * 32, 32);
    v16bf b0 = load_bt(Bs + nc * 32,        32);
    v16bf b1 = load_bt(Bs + (nc + 16) * 32, 32);
    c00 = wmma_bf16(a0, b0, c00);
    c01 = wmma_bf16(a0, b1, c01);
    c10 = wmma_bf16(a1, b0, c10);
    c11 = wmma_bf16(a1, b1, c11);
  }

  // ---- epilogue ----
  int half = lane >> 4, n = lane & 15;
  int col0 = n0 + nc + n, col1 = col0 + 16;
  float bb0 = bias ? bias[col0] : 0.0f;
  float bb1 = bias ? bias[col1] : 0.0f;

#pragma unroll
  for (int mt = 0; mt < 2; ++mt) {
    v8f cA = mt ? c10 : c00;
    v8f cB = mt ? c11 : c01;
    int mbase = m0 + mr + mt * 16;
    if (OMODE == 2) {
      // bf16 store in [B,NH,DH,SEQ]; lane's 8 rows are contiguous tokens
      v8bf o0, o1;
#pragma unroll
      for (int r = 0; r < 8; ++r) {
        o0[r] = (__bf16)((cA[r] + bb0) * scale);
        o1[r] = (__bf16)((cB[r] + bb1) * scale);
      }
      int mm = mbase + 8 * half;  // first token of chunk (mult of 8)
      int bi = mm >> 10, tok = mm & (SEQ - 1);
      __bf16* O = (__bf16*)Outp;
      int h0 = col0 >> 6, dh0 = col0 & 63;
      int h1 = col1 >> 6, dh1 = col1 & 63;
      *(v8bf*)(O + ((size_t)((bi * NH + h0) * DH + dh0)) * SEQ + tok) = o0;
      *(v8bf*)(O + ((size_t)((bi * NH + h1) * DH + dh1)) * SEQ + tok) = o1;
    } else {
#pragma unroll
      for (int r = 0; r < 8; ++r) {
        int m = mbase + r + 8 * half;
        float o0 = (cA[r] + bb0) * scale;
        float o1 = (cB[r] + bb1) * scale;
        if (OMODE == 1) {
          float* O = (float*)Outp;
          O[(size_t)m * Nc + col0] = o0;
          O[(size_t)m * Nc + col1] = o1;
        } else {
          __bf16* O = (__bf16*)Outp;
          O[(size_t)m * Nc + col0] = (__bf16)o0;
          O[(size_t)m * Nc + col1] = (__bf16)o1;
        }
      }
    }
  }
}

// ---------------------------------------------------------------------------
// Attention: one block = (b, h, 64-query stripe), 8 waves.
// Phase 1: dual-path QK^T via WMMA -> 64x1024 fp32 logit stripe in LDS
// Phase 2: exact softmax (float4-vectorized), emit fp32 w
// Phase 3: W @ V via WMMA (A = probs fp32->bf16 on load; B = vT tiles)
// ---------------------------------------------------------------------------
__global__ __launch_bounds__(256) void attn_kernel(
    const __bf16* __restrict__ q, const __bf16* __restrict__ qcs,
    const __bf16* __restrict__ kmat, const __bf16* __restrict__ kcmat,
    const __bf16* __restrict__ vT, const float* __restrict__ mask,
    const float* __restrict__ gate, float* __restrict__ w_out,
    __bf16* __restrict__ wv_out) {
  extern __shared__ __align__(32) char smem[];
  float*  logits = (float*)smem;                                     // QT x SEQ
  __bf16* qa  = (__bf16*)(smem + (size_t)QT * SEQ * sizeof(float));  // [row][64]
  __bf16* qca = qa  + QT * DH;                                       // [row][64]
  __bf16* ktA = qca + QT * DH;  // Bt for QK: [key][dh]  (natural K layout)
  __bf16* kct = ktA + 64 * DH;  // [key][dh]
  __bf16* vs  = kct + 64 * DH;  // Bt for WV: [dh][32 keys]
  __shared__ float redp[256];
  __shared__ float rowv[QT];

  const int tid  = threadIdx.x;
  const int lane = tid & 31;
  const int wave = tid >> 5;
  const int qt = blockIdx.x;  // 0..15
  const int h  = blockIdx.y;  // 0..11
  const int b  = blockIdx.z;  // 0..3
  const int qbase = qt * QT;

  const float g  = 1.0f / (1.0f + expf(-gate[h]));
  const float gm = 1.0f - g;

  // stage q / q_cs stripes (32B vectors)
  {
    int row = tid >> 2, db = (tid & 3) * 16;
    size_t base = ((size_t)(b * SEQ + qbase + row)) * DM + h * DH + db;
    *(v16bf*)(qa  + row * DH + db) = *(const v16bf*)(q   + base);
    *(v16bf*)(qca + row * DH + db) = *(const v16bf*)(qcs + base);
  }
  __syncthreads();

  const int mr = (wave >> 1) * 16;  // query rows within stripe
  const int nc = (wave & 1) * 32;   // key cols within 64-key tile

  // hoisted loop-invariant A fragments
  v16bf a0  = load_a_bf(qa  + mr * DH,      DH);
  v16bf a1  = load_a_bf(qa  + mr * DH + 32, DH);
  v16bf ac0 = load_a_bf(qca + mr * DH,      DH);
  v16bf ac1 = load_a_bf(qca + mr * DH + 32, DH);

  // ----- phase 1: blended logits -----
  for (int ktile = 0; ktile < SEQ / 64; ++ktile) {
    __syncthreads();
    {  // stage K / K_cs tiles in natural [key][dh] layout (= Bt for QK)
      int key = tid >> 2, db = (tid & 3) * 16;
      size_t base = ((size_t)(b * SEQ + ktile * 64 + key)) * DM + h * DH + db;
      *(v16bf*)(ktA + key * DH + db) = *(const v16bf*)(kmat  + base);
      *(v16bf*)(kct + key * DH + db) = *(const v16bf*)(kcmat + base);
    }
    __syncthreads();
#pragma unroll
    for (int t = 0; t < 2; ++t) {
      int cc = nc + t * 16;  // key-column group
      v16bf b0 = load_bt(ktA + cc * DH,      DH);
      v16bf b1 = load_bt(ktA + cc * DH + 32, DH);
      v8f s = {};
      s = wmma_bf16(a0, b0, s);
      s = wmma_bf16(a1, b1, s);
      v16bf bc0 = load_bt(kct + cc * DH,      DH);
      v16bf bc1 = load_bt(kct + cc * DH + 32, DH);
      v8f scs = {};
      scs = wmma_bf16(ac0, bc0, scs);
      scs = wmma_bf16(ac1, bc1, scs);
      int half = lane >> 4, n = lane & 15;
#pragma unroll
      for (int r = 0; r < 8; ++r) {
        int m   = mr + r + 8 * half;
        int key = ktile * 64 + cc + n;
        logits[m * SEQ + key] =
            gm * s[r] + g * scs[r] + mask[(size_t)(qbase + m) * SEQ + key];
      }
    }
  }
  __syncthreads();

  // ----- phase 2: exact softmax over full rows (float4 vectorized) -----
  {
    int row = tid >> 2, p = tid & 3;
    float* lrow = logits + row * SEQ;
    float mx = -3.0e38f;
    for (int c = p * 4; c < SEQ; c += 16) {
      float4 t = *(float4*)(lrow + c);
      mx = fmaxf(fmaxf(fmaxf(mx, t.x), t.y), fmaxf(t.z, t.w));
    }
    redp[tid] = mx;
    __syncthreads();
    if (p == 0)
      rowv[row] = fmaxf(fmaxf(redp[tid], redp[tid + 1]),
                        fmaxf(redp[tid + 2], redp[tid + 3]));
    __syncthreads();
    float rm = rowv[row];
    float sm = 0.0f;
    for (int c = p * 4; c < SEQ; c += 16) {
      float4 t = *(float4*)(lrow + c);
      t.x = __expf(t.x - rm);
      t.y = __expf(t.y - rm);
      t.z = __expf(t.z - rm);
      t.w = __expf(t.w - rm);
      *(float4*)(lrow + c) = t;
      sm += t.x + t.y + t.z + t.w;
    }
    redp[tid] = sm;
    __syncthreads();
    if (p == 0)
      rowv[row] = redp[tid] + redp[tid + 1] + redp[tid + 2] + redp[tid + 3];
    __syncthreads();
    float inv = 1.0f / rowv[row];
    float* wrow = w_out + ((size_t)((b * NH + h) * SEQ) + qbase + row) * SEQ;
    for (int c = p * 4; c < SEQ; c += 16) {
      float4 t = *(float4*)(lrow + c);
      t.x *= inv; t.y *= inv; t.z *= inv; t.w *= inv;
      *(float4*)(lrow + c) = t;
      *(float4*)(wrow + c) = t;
    }
  }

  // ----- phase 3: wv = w @ v  (B from vT[B,NH,DH,SEQ]) -----
  v8f c0 = {}, c1 = {};
  const __bf16* vhead = vT + ((size_t)(b * NH + h)) * DH * SEQ;
  for (int ks = 0; ks < SEQ / 32; ++ks) {
    __syncthreads();
    {  // stage vT tile [64 dh][32 keys] — vectorized copy
      int dh = tid >> 2, kb = (tid & 3) * 8;
      *(v8bf*)(vs + dh * 32 + kb) =
          *(const v8bf*)(vhead + (size_t)dh * SEQ + ks * 32 + kb);
    }
    __syncthreads();
    v16bf a  = load_a_f32(logits + mr * SEQ + ks * 32, SEQ);
    v16bf b0 = load_bt(vs + nc * 32,        32);
    v16bf b1 = load_bt(vs + (nc + 16) * 32, 32);
    c0 = wmma_bf16(a, b0, c0);
    c1 = wmma_bf16(a, b1, c1);
  }
  {  // store wv in [B,N,D] layout (merged heads)
    int half = lane >> 4, n = lane & 15;
#pragma unroll
    for (int r = 0; r < 8; ++r) {
      int m = qbase + mr + r + 8 * half;
      size_t base = ((size_t)(b * SEQ + m)) * DM + h * DH + nc;
      wv_out[base + n]      = (__bf16)c0[r];
      wv_out[base + 16 + n] = (__bf16)c1[r];
    }
  }
}

// ---------------------------------------------------------------------------
extern "C" void kernel_launch(void* const* d_in, const int* in_sizes, int n_in,
                              void* d_out, int out_size, void* d_ws,
                              size_t ws_size, hipStream_t stream) {
  const float* x    = (const float*)d_in[0];
  const float* mask = (const float*)d_in[1];
  const float* Wq   = (const float*)d_in[2];
  const float* bq   = (const float*)d_in[3];
  const float* Wk   = (const float*)d_in[4];
  const float* Wv   = (const float*)d_in[5];
  const float* bv   = (const float*)d_in[6];
  const float* Wo   = (const float*)d_in[7];
  const float* bo   = (const float*)d_in[8];
  const float* Wqcs = (const float*)d_in[9];
  const float* bqcs = (const float*)d_in[10];
  const float* Wkcs = (const float*)d_in[11];
  const float* gate = (const float*)d_in[12];

  char* ws = (char*)d_ws;
  const size_t projBytes = (size_t)MTOT * DM * 2;  // bf16
  __bf16* q   = (__bf16*)(ws);
  __bf16* qcs = (__bf16*)(ws + 1 * projBytes);
  __bf16* k   = (__bf16*)(ws + 2 * projBytes);
  __bf16* kcs = (__bf16*)(ws + 3 * projBytes);
  __bf16* vT  = (__bf16*)(ws + 4 * projBytes);  // [B,NH,DH,SEQ]
  __bf16* wv  = (__bf16*)(ws + 5 * projBytes);

  const float scale = 0.3535533905932738f;  // DH^-0.25

  dim3 gg(MTOT / 128, DM / 64), blk(256);
  gemm_bt_kernel<0, 0><<<gg, blk, 0, stream>>>(x, Wq,   bq,      scale, q,   MTOT, DM, DM);
  gemm_bt_kernel<0, 0><<<gg, blk, 0, stream>>>(x, Wqcs, bqcs,    scale, qcs, MTOT, DM, DM);
  gemm_bt_kernel<0, 0><<<gg, blk, 0, stream>>>(x, Wk,   nullptr, scale, k,   MTOT, DM, DM);
  gemm_bt_kernel<0, 0><<<gg, blk, 0, stream>>>(x, Wkcs, nullptr, scale, kcs, MTOT, DM, DM);
  gemm_bt_kernel<0, 2><<<gg, blk, 0, stream>>>(x, Wv,   bv,      1.0f,  vT,  MTOT, DM, DM);

  float* outp  = (float*)d_out;
  float* w_out = outp + (size_t)MTOT * DM;

  size_t lds = (size_t)QT * SEQ * sizeof(float) +
               (size_t)(2 * QT * DH + 2 * 64 * DH + 32 * DH) * 2;
  attn_kernel<<<dim3(SEQ / QT, NH, Bsz), blk, lds, stream>>>(
      q, qcs, k, kcs, vT, mask, gate, w_out, wv);

  gemm_bt_kernel<1, 1><<<gg, blk, 0, stream>>>(wv, Wo, bo, 1.0f, outp, MTOT, DM, DM);
}
